// CREN_81973745811734
// MI455X (gfx1250) — matrix operand order repaired and checked by reading, not compiled, not run on hardware
//
#include <hip/hip_runtime.h>
#include <hip/hip_bf16.h>
#include <math.h>

// ---------------------------------------------------------------------------
// CREN forward for MI455X (gfx1250, wave32, WMMA).
//
// Roofline: ~60 GFLOP vs ~128MB HBM traffic (x in, x_dot out; model matrices
// < 2MB -> L2-resident) => ~460 FLOP/byte, compute bound. All large GEMMs
// (x@C1^T, w@D11_blk^T, x@A^T + w@B1^T) run on v_wmma_f32_16x16x32_bf16.
// Output-column register blocking (4 accumulator tiles/wave) amortizes each
// LDS A-fragment across 4 WMMAs: LDS A-traffic ~128B/clk < 256B/clk WGP LDS
// bandwidth, so the matrix pipes, not LDS, are the limiter.
// The sequential tanh recurrence and the P^-1 solve stay in fp32.
// ---------------------------------------------------------------------------

#define DX 256
#define DV 512
#define TM 64       // rows per workgroup in the main kernel

typedef __attribute__((ext_vector_type(16))) __bf16 v16bf;
typedef __attribute__((ext_vector_type(8)))  __bf16 v8bf;
typedef __attribute__((ext_vector_type(8)))  float  v8f;

// ---------------------------------------------------------------------------
// Small setup kernels
// ---------------------------------------------------------------------------

// out = alpha * M @ M^T + eps * I   (M is n x n, row-major)
__global__ void k_syrk(float* __restrict__ out, const float* __restrict__ M,
                       int n, float alpha, float eps) {
  int e = blockIdx.x * blockDim.x + threadIdx.x;
  if (e >= n * n) return;
  int i = e / n, j = e % n;
  const float* ri = M + (size_t)i * n;
  const float* rj = M + (size_t)j * n;
  float s = 0.f;
  for (int k = 0; k < n; ++k) s += ri[k] * rj[k];
  out[e] = alpha * s + (i == j ? eps : 0.f);
}

// lamInv[i] = 1 / (0.5 * H4[i][i]) = 2 / H[(256+i)*768 + 256+i]
__global__ void k_laminv(float* __restrict__ lamInv, const float* __restrict__ H) {
  int i = blockIdx.x * blockDim.x + threadIdx.x;
  if (i < DV) lamInv[i] = 2.0f / H[(size_t)(DX + i) * 768 + DX + i];
}

// W = [ Y | -H2 - Chi ]  (256 x 768),  Y = -0.5*(H1 + Y1 - Y1^T)
__global__ void k_buildW(float* __restrict__ W, const float* __restrict__ H,
                         const float* __restrict__ Y1, const float* __restrict__ Chi) {
  int e = blockIdx.x * blockDim.x + threadIdx.x;
  if (e >= DX * 768) return;
  int r = e / 768, c = e % 768;
  float v;
  if (c < DX) {
    v = -0.5f * (H[(size_t)r * 768 + c] + Y1[(size_t)r * DX + c] - Y1[(size_t)c * DX + r]);
  } else {
    int vv = c - DX;
    v = -H[(size_t)r * 768 + c] - Chi[(size_t)r * DV + vv];
  }
  W[e] = v;
}

// D11[i][j] = (j<i) ? -H4[i][j]/lam[i] : 0   -> fp32 (serial path) + bf16 (WMMA B)
__global__ void k_buildD11(float* __restrict__ D11f, __bf16* __restrict__ D11bf,
                           const float* __restrict__ H, const float* __restrict__ lamInv) {
  int e = blockIdx.x * blockDim.x + threadIdx.x;
  if (e >= DV * DV) return;
  int i = e >> 9, j = e & (DV - 1);
  float v = 0.f;
  if (j < i) v = -H[(size_t)(DX + i) * 768 + DX + j] * lamInv[i];
  D11f[e] = v;
  D11bf[e] = (__bf16)v;
}

// C1[v][k] = Chi[k][v] / lam[v]  (row-major 512x256, bf16)
__global__ void k_buildC1(__bf16* __restrict__ C1bf, const float* __restrict__ Chi,
                          const float* __restrict__ lamInv) {
  int e = blockIdx.x * blockDim.x + threadIdx.x;
  if (e >= DV * DX) return;
  int v = e >> 8, k = e & (DX - 1);
  C1bf[e] = (__bf16)(Chi[(size_t)k * DV + v] * lamInv[v]);
}

// ---------------------------------------------------------------------------
// In-LDS LU factorization of P (256x256, SPD, no pivoting). Single workgroup.
// LDS = 256KB (fits in the 320KB WGP LDS).
// ---------------------------------------------------------------------------
__global__ __launch_bounds__(1024) void k_lu(float* __restrict__ LUg,
                                             const float* __restrict__ Pg) {
  extern __shared__ float Pl[];  // 256*256
  const int tid = threadIdx.x;
  for (int e = tid; e < DX * DX; e += 1024) Pl[e] = Pg[e];
  __syncthreads();
  for (int k = 0; k < DX; ++k) {
    float invp = 1.0f / Pl[k * DX + k];      // broadcast LDS read
    for (int i = k + 1 + tid; i < DX; i += 1024) Pl[i * DX + k] *= invp;
    __syncthreads();
    int rg = tid >> 2, cp = tid & 3;
    for (int i = k + 1 + rg; i < DX; i += 256) {
      float lik = Pl[i * DX + k];
      for (int j = k + 1 + cp; j < DX; j += 4) Pl[i * DX + j] -= lik * Pl[k * DX + j];
    }
    __syncthreads();
  }
  for (int e = tid; e < DX * DX; e += 1024) LUg[e] = Pl[e];
}

// ---------------------------------------------------------------------------
// Triangular solve Z = P^-1 W for 768 columns, one column per wave, column in
// VGPRs (8 regs/lane), pivot broadcast via __shfl (wave32 lockstep => no
// barriers in the k-loop). LU kept TRANSPOSED in LDS so the per-step gather
// of column k is a contiguous LDS row (bank-conflict free).
// Emits directly to Abf (cols 0..255) and B1bf (cols 256..767) in bf16.
// ---------------------------------------------------------------------------
__global__ __launch_bounds__(1024) void k_solve(__bf16* __restrict__ Abf,
                                                __bf16* __restrict__ B1bf,
                                                const float* __restrict__ LUg,
                                                const float* __restrict__ W) {
  extern __shared__ float LUt[];  // LUt[k*256+i] = LU[i][k]
  const int tid = threadIdx.x, wave = tid >> 5, lane = tid & 31;
  for (int e = tid; e < DX * DX; e += 1024) {
    int i = e >> 8, k = e & (DX - 1);
    LUt[k * DX + i] = LUg[e];
  }
  __syncthreads();

  const int col = blockIdx.x * 32 + wave;  // 24 blocks * 32 waves = 768 cols
  float z[8];
#pragma unroll
  for (int g = 0; g < 8; ++g) z[g] = W[(size_t)(lane + 32 * g) * 768 + col];

  // forward: L y = b (unit diag)
#pragma unroll
  for (int g = 0; g < 8; ++g) {
    for (int kk = 0; kk < 32; ++kk) {
      int k = g * 32 + kk;
      float xk = __shfl(z[g], kk);
#pragma unroll
      for (int gg = 0; gg < 8; ++gg) {
        int row = lane + 32 * gg;
        if (row > k) z[gg] -= LUt[k * DX + row] * xk;
      }
    }
  }
  // backward: U x = y
#pragma unroll
  for (int g = 7; g >= 0; --g) {
    for (int kk = 31; kk >= 0; --kk) {
      int k = g * 32 + kk;
      float xk = __shfl(z[g], kk) / LUt[k * DX + k];
      if (lane == kk) z[g] = xk;
#pragma unroll
      for (int gg = 0; gg < 8; ++gg) {
        int row = lane + 32 * gg;
        if (row < k) z[gg] -= LUt[k * DX + row] * xk;
      }
    }
  }
#pragma unroll
  for (int g = 0; g < 8; ++g) {
    int row = lane + 32 * g;
    if (col < DX) Abf[(size_t)row * DX + col] = (__bf16)z[g];
    else          B1bf[(size_t)row * DV + (col - DX)] = (__bf16)z[g];
  }
}

// ---------------------------------------------------------------------------
// Main fused kernel: per 64-row tile of x:
//   1. x tile -> LDS bf16 (float4-vectorized staging)
//   2. cx = x @ C1^T + bv            (WMMA, 4-wide column register blocking)
//   3. 32 column-blocks of 16: inter-block GEMM w[:, :16b] @ D11_blk^T via
//      WMMA (K padded to 32; padded w columns are still zero), then the 16
//      serial tanh steps in fp32 (rows -> lanes, diag block in LDS)
//   4. x_dot = x @ A^T + w @ B1^T + bx  (WMMA, 4-wide column blocking)
// w never leaves LDS. LDS use ~233KB -> 1 workgroup (8 waves) per WGP.
// ---------------------------------------------------------------------------
__global__ __launch_bounds__(256) void k_main(const float* __restrict__ x,
    const __bf16* __restrict__ C1bf, const __bf16* __restrict__ D11bf,
    const float* __restrict__ D11f, const __bf16* __restrict__ Abf,
    const __bf16* __restrict__ B1bf, const float* __restrict__ bv,
    const float* __restrict__ bx, float* __restrict__ out) {
  extern __shared__ char smem[];
  __bf16* xt  = (__bf16*)smem;                              // 64*256 bf16 (32KB)
  __bf16* wt  = (__bf16*)(smem + 32768);                    // 64*512 bf16 (64KB)
  float*  cx  = (float*)(smem + 32768 + 65536);             // 64*512 f32 (128KB)
  float*  accA = (float*)(smem + 32768 + 65536 + 131072);   // 64*16 f32
  float*  accB = accA + TM * 16;                            // 64*16 f32
  float*  dblk = accB + TM * 16;                            // 16*16 f32

  const int tid = threadIdx.x, wave = tid >> 5, lane = tid & 31;
  const int hi = lane >> 4, l16 = lane & 15, mb = hi * 8;
  const int rg = wave & 3, half = wave >> 2;   // row group / column half
  const int m0 = rg * 16;
  const int r0 = blockIdx.x * TM;

  // Phase 1: stage x tile as bf16 (float4 loads), zero w tile.
  for (int e = tid * 4; e < TM * DX; e += 256 * 4) {
    int r = e >> 8, c = e & (DX - 1);
    float4 v = *(const float4*)(x + (size_t)(r0 + r) * DX + c);
    xt[e + 0] = (__bf16)v.x; xt[e + 1] = (__bf16)v.y;
    xt[e + 2] = (__bf16)v.z; xt[e + 3] = (__bf16)v.w;
  }
  for (int e = tid; e < TM * DV; e += 256) wt[e] = (__bf16)0.f;
  __syncthreads();

  // A operand (16-bit 16x32): lane<16 holds M=l16, K = k0+{0..15};
  // lane>=16 holds K = k0+{16..31}. Two 16B LDS loads.
  auto loadA = [&](const __bf16* base, int m, int ld, int k0) -> v16bf {
    const __bf16* p = base + m * ld + k0 + hi * 16;
    v8bf lo = *(const v8bf*)p;
    v8bf hv = *(const v8bf*)(p + 8);
    return __builtin_shufflevector(lo, hv, 0, 1, 2, 3, 4, 5, 6, 7,
                                           8, 9, 10, 11, 12, 13, 14, 15);
  };
  // B operand (16-bit 32x16): lane holds column n = n0+l16, K-run
  // k0 + hi*16 .. +15 => contiguous 32B from row n of the row-major weight.
  auto loadB = [&](const __bf16* base, int n, int ld, int k0) -> v16bf {
    return *(const v16bf*)(base + (size_t)n * ld + k0 + hi * 16);
  };

  // Phase 2: cx = xt @ C1^T + bv.  Each wave: fixed row group, 16 column
  // tiles in 4 register-blocked groups of 4 (A fragment reused 4x per load).
  for (int blk = 0; blk < 4; ++blk) {
    int cg0 = half * 16 + blk * 4;
    v8f acc0 = {0.f,0.f,0.f,0.f,0.f,0.f,0.f,0.f};
    v8f acc1 = acc0, acc2 = acc0, acc3 = acc0;
    for (int k0 = 0; k0 < DX; k0 += 32) {
      __builtin_prefetch(C1bf + (size_t)(cg0 * 16 + l16) * DX + k0 + 64, 0, 1);
      v16bf a = loadA(xt, m0 + l16, DX, k0);
      v16bf b0 = loadB(C1bf, (cg0 + 0) * 16 + l16, DX, k0);
      v16bf b1 = loadB(C1bf, (cg0 + 1) * 16 + l16, DX, k0);
      v16bf b2 = loadB(C1bf, (cg0 + 2) * 16 + l16, DX, k0);
      v16bf b3 = loadB(C1bf, (cg0 + 3) * 16 + l16, DX, k0);
      acc0 = __builtin_amdgcn_wmma_f32_16x16x32_bf16(false, a, false, b0, (short)0, acc0, false, false);
      acc1 = __builtin_amdgcn_wmma_f32_16x16x32_bf16(false, a, false, b1, (short)0, acc1, false, false);
      acc2 = __builtin_amdgcn_wmma_f32_16x16x32_bf16(false, a, false, b2, (short)0, acc2, false, false);
      acc3 = __builtin_amdgcn_wmma_f32_16x16x32_bf16(false, a, false, b3, (short)0, acc3, false, false);
    }
#pragma unroll
    for (int c = 0; c < 4; ++c) {
      const v8f& acc = c == 0 ? acc0 : c == 1 ? acc1 : c == 2 ? acc2 : acc3;
      int v0 = (cg0 + c) * 16;
      float bvv = bv[v0 + l16];
#pragma unroll
      for (int g = 0; g < 8; ++g) cx[(m0 + mb + g) * DV + v0 + l16] = acc[g] + bvv;
    }
  }
  __syncthreads();

  // Phase 3: blocked triangular tanh recurrence.
  for (int b = 0; b < DV / 16; ++b) {
    int nch = (b + 1) >> 1;              // K chunks of 32 covering [0,16b) + zero pad
    int cLo = (half == 0) ? 0 : (nch + 1) >> 1;   // K split across wave pairs
    int cHi = (half == 0) ? (nch + 1) >> 1 : nch;
    v8f acc = {0.f,0.f,0.f,0.f,0.f,0.f,0.f,0.f};
    for (int c = cLo; c < cHi; ++c) {
      int k0 = c * 32;
      v16bf a = loadA(wt, m0 + l16, DV, k0);
      v16bf bb = loadB(D11bf, 16 * b + l16, DV, k0);
      acc = __builtin_amdgcn_wmma_f32_16x16x32_bf16(false, a, false, bb,
                                                    (short)0, acc, false, false);
    }
    float* accP = (half == 0) ? accA : accB;
#pragma unroll
    for (int g = 0; g < 8; ++g) accP[(m0 + mb + g) * 16 + l16] = acc[g];
    // diag 16x16 block of D11 in fp32 for the serial part
    dblk[tid] = D11f[(size_t)(16 * b + (tid >> 4)) * DV + 16 * b + (tid & 15)];
    __syncthreads();

    if (tid < TM) {  // 2 waves: lane <-> row, 16 serial tanh steps in fp32
      int r = tid;
      const float* cxr = cx + r * DV + 16 * b;
      const float* aAr = accA + r * 16;
      const float* aBr = accB + r * 16;
      float wreg[16];
      v16bf wv;
#pragma unroll
      for (int i = 0; i < 16; ++i) {
        float v = cxr[i] + aAr[i] + aBr[i];
        for (int j = 0; j < i; ++j) v += dblk[i * 16 + j] * wreg[j];
        wreg[i] = tanhf(v);
        wv[i] = (__bf16)wreg[i];
      }
      *(v16bf*)(wt + r * DV + 16 * b) = wv;  // 32B contiguous store
    }
    __syncthreads();
  }

  // Phase 4: out = xt @ A^T + wt @ B1^T + bx.  Each wave: fixed row group,
  // 8 column tiles in 2 register-blocked groups of 4.
  for (int blk = 0; blk < 2; ++blk) {
    int cg0 = half * 8 + blk * 4;
    v8f acc0 = {0.f,0.f,0.f,0.f,0.f,0.f,0.f,0.f};
    v8f acc1 = acc0, acc2 = acc0, acc3 = acc0;
    for (int k0 = 0; k0 < DX; k0 += 32) {
      v16bf a = loadA(xt, m0 + l16, DX, k0);
      v16bf b0 = loadB(Abf, (cg0 + 0) * 16 + l16, DX, k0);
      v16bf b1 = loadB(Abf, (cg0 + 1) * 16 + l16, DX, k0);
      v16bf b2 = loadB(Abf, (cg0 + 2) * 16 + l16, DX, k0);
      v16bf b3 = loadB(Abf, (cg0 + 3) * 16 + l16, DX, k0);
      acc0 = __builtin_amdgcn_wmma_f32_16x16x32_bf16(false, a, false, b0, (short)0, acc0, false, false);
      acc1 = __builtin_amdgcn_wmma_f32_16x16x32_bf16(false, a, false, b1, (short)0, acc1, false, false);
      acc2 = __builtin_amdgcn_wmma_f32_16x16x32_bf16(false, a, false, b2, (short)0, acc2, false, false);
      acc3 = __builtin_amdgcn_wmma_f32_16x16x32_bf16(false, a, false, b3, (short)0, acc3, false, false);
    }
    for (int k0 = 0; k0 < DV; k0 += 32) {
      v16bf a = loadA(wt, m0 + l16, DV, k0);
      v16bf b0 = loadB(B1bf, (cg0 + 0) * 16 + l16, DV, k0);
      v16bf b1 = loadB(B1bf, (cg0 + 1) * 16 + l16, DV, k0);
      v16bf b2 = loadB(B1bf, (cg0 + 2) * 16 + l16, DV, k0);
      v16bf b3 = loadB(B1bf, (cg0 + 3) * 16 + l16, DV, k0);
      acc0 = __builtin_amdgcn_wmma_f32_16x16x32_bf16(false, a, false, b0, (short)0, acc0, false, false);
      acc1 = __builtin_amdgcn_wmma_f32_16x16x32_bf16(false, a, false, b1, (short)0, acc1, false, false);
      acc2 = __builtin_amdgcn_wmma_f32_16x16x32_bf16(false, a, false, b2, (short)0, acc2, false, false);
      acc3 = __builtin_amdgcn_wmma_f32_16x16x32_bf16(false, a, false, b3, (short)0, acc3, false, false);
    }
#pragma unroll
    for (int c = 0; c < 4; ++c) {
      const v8f& acc = c == 0 ? acc0 : c == 1 ? acc1 : c == 2 ? acc2 : acc3;
      int c0 = (cg0 + c) * 16;
      float bxx = bx[c0 + l16];
#pragma unroll
      for (int g = 0; g < 8; ++g)
        out[(size_t)(r0 + m0 + mb + g) * DX + c0 + l16] = acc[g] + bxx;
    }
  }
}

// ---------------------------------------------------------------------------
// Host launch
// ---------------------------------------------------------------------------
extern "C" void kernel_launch(void* const* d_in, const int* in_sizes, int n_in,
                              void* d_out, int out_size, void* d_ws, size_t ws_size,
                              hipStream_t stream) {
  const float* x     = (const float*)d_in[1];
  const float* Pstar = (const float*)d_in[2];
  const float* Chi   = (const float*)d_in[3];
  const float* Xm    = (const float*)d_in[4];
  const float* Y1    = (const float*)d_in[5];
  const float* bv    = (const float*)d_in[8];
  const float* bx    = (const float*)d_in[9];
  float* out = (float*)d_out;
  const int N = in_sizes[1] / DX;
  const float EPS = 0.05f;

  // workspace carve-up (~6MB; L2-resident)
  char* p = (char*)d_ws;
  auto take = [&](size_t bytes) { char* r = p; p += (bytes + 255) & ~size_t(255); return r; };
  float*  H      = (float*)take(768 * 768 * 4);
  float*  Pm     = (float*)take(DX * DX * 4);
  float*  LU     = (float*)take(DX * DX * 4);
  float*  W      = (float*)take(DX * 768 * 4);
  float*  lamInv = (float*)take(DV * 4);
  float*  D11f   = (float*)take(DV * DV * 4);
  __bf16* Abf    = (__bf16*)take(DX * DX * 2);
  __bf16* B1bf   = (__bf16*)take(DX * DV * 2);
  __bf16* C1bf   = (__bf16*)take(DV * DX * 2);
  __bf16* D11bf  = (__bf16*)take(DV * DV * 2);
  (void)ws_size; (void)n_in; (void)out_size;

  k_syrk<<<(768 * 768 + 255) / 256, 256, 0, stream>>>(H, Xm, 768, 1.0f, EPS);
  k_syrk<<<(DX * DX + 255) / 256, 256, 0, stream>>>(Pm, Pstar, DX, 0.5f, EPS);
  k_laminv<<<(DV + 255) / 256, 256, 0, stream>>>(lamInv, H);
  k_buildW<<<(DX * 768 + 255) / 256, 256, 0, stream>>>(W, H, Y1, Chi);
  k_buildD11<<<(DV * DV + 255) / 256, 256, 0, stream>>>(D11f, D11bf, H, lamInv);
  k_buildC1<<<(DV * DX + 255) / 256, 256, 0, stream>>>(C1bf, Chi, lamInv);
  k_lu<<<1, 1024, DX * DX * 4, stream>>>(LU, Pm);
  k_solve<<<768 / 32, 1024, DX * DX * 4, stream>>>(Abf, B1bf, LU, W);

  const size_t mainLds = 32768 + 65536 + 131072 + 2 * TM * 16 * 4 + 16 * 16 * 4;
  k_main<<<(N + TM - 1) / TM, 256, mainLds, stream>>>(
      x, C1bf, D11bf, D11f, Abf, B1bf, bv, bx, out);
}
